// Decoder_51823075393921
// MI455X (gfx1250) — compile-verified, matrix operand order checked by reading, not run on hardware
//
#include <hip/hip_runtime.h>
#include <hip/hip_bf16.h>
#include <math.h>

#define B_    128
#define S_    400
#define E_    512
#define H_    512
#define V_    50257
#define VEXT_ 50307

typedef __attribute__((ext_vector_type(2)))  float        v2f;
typedef __attribute__((ext_vector_type(8)))  float        v8f;
typedef __attribute__((ext_vector_type(8)))  __bf16       v8bf;
typedef __attribute__((ext_vector_type(16))) __bf16       v16bf;
typedef __attribute__((ext_vector_type(4)))  unsigned int v4u;
typedef __attribute__((ext_vector_type(4)))  int          v4i;
typedef __attribute__((ext_vector_type(8)))  int          v8i;

// ---------------------------------------------------------------------------
// f32 WMMA GEMM:  C[128,N] = A1[128,K1] @ B1^T + A2[128,K2] @ B2^T + bias1 + bias2
// One wave per 16-column tile; wave carries all 8 M-tiles (M=128) so the B
// fragment is loaded once per K-step and reused 8x.
// ---------------------------------------------------------------------------
__global__ __launch_bounds__(32) void wmma_gemm_f32(
    const float* __restrict__ A1, const float* __restrict__ B1, int ldb1, int K1,
    const float* __restrict__ A2, const float* __restrict__ B2, int ldb2, int K2,
    const float* __restrict__ bias1, const float* __restrict__ bias2,
    float* __restrict__ C, int N)
{
  const int lane = threadIdx.x;
  const int hf   = lane >> 4;       // half-wave select
  const int m16  = lane & 15;
  const int n    = blockIdx.x * 16 + m16;
  const int nc   = (n < N) ? n : (N - 1);
  const float bs = (n < N) ? 1.0f : 0.0f;

  v8f acc[8] = {};

  for (int k = 0; k < K1; k += 4) {
    const float* bp = B1 + (size_t)nc * ldb1 + k + 2 * hf;
    v2f bf; bf[0] = bp[0] * bs; bf[1] = bp[1] * bs;
#pragma unroll
    for (int t = 0; t < 8; ++t) {
      const float* ap = A1 + (size_t)(t * 16 + m16) * K1 + k + 2 * hf;
      v2f af; af[0] = ap[0]; af[1] = ap[1];
      acc[t] = __builtin_amdgcn_wmma_f32_16x16x4_f32(false, af, false, bf,
                                                     (short)0, acc[t], false, false);
    }
  }
  if (A2) {
    for (int k = 0; k < K2; k += 4) {
      const float* bp = B2 + (size_t)nc * ldb2 + k + 2 * hf;
      v2f bf; bf[0] = bp[0] * bs; bf[1] = bp[1] * bs;
#pragma unroll
      for (int t = 0; t < 8; ++t) {
        const float* ap = A2 + (size_t)(t * 16 + m16) * K2 + k + 2 * hf;
        v2f af; af[0] = ap[0]; af[1] = ap[1];
        acc[t] = __builtin_amdgcn_wmma_f32_16x16x4_f32(false, af, false, bf,
                                                       (short)0, acc[t], false, false);
      }
    }
  }
  if (n < N) {
    const float bsum = (bias1 ? bias1[n] : 0.0f) + (bias2 ? bias2[n] : 0.0f);
#pragma unroll
    for (int t = 0; t < 8; ++t)
#pragma unroll
      for (int v = 0; v < 8; ++v) {
        const int m = t * 16 + v + 8 * hf;
        C[(size_t)m * N + n] = acc[t][v] + bsum;
      }
  }
}

// ---------------------------------------------------------------------------
// fp32 -> bf16 conversion (A operand of the vocab GEMM; 256KB -> 128KB, L2
// resident for all subsequent blocks).
// ---------------------------------------------------------------------------
__global__ void cvt_bf16_kernel(const float* __restrict__ in,
                                __bf16* __restrict__ out, int n)
{
  const int i = blockIdx.x * blockDim.x + threadIdx.x;
  if (i < n) out[i] = (__bf16)in[i];
}

// ---------------------------------------------------------------------------
// bf16 WMMA GEMM with TDM-staged B:  C[128,N] = A[128,K]bf16 @ Bt[N,K]^T + bias
// One wave per 16-column tile. The wave's 16xK fp32 slab of Bt is DMA'd into
// LDS by the Tensor Data Mover (one tensor_load_to_lds per block):
//   - rows beyond N return zero (TDM OOB) -> ragged last tile needs no clamps
//   - pad_enable inserts 2 DWORDs every 256 -> row stride 516 floats
//     => 16 rows land on distinct LDS banks (stride 4 banks)
// A is pre-converted bf16; each fragment is two contiguous 16B loads.
// ---------------------------------------------------------------------------
__global__ __launch_bounds__(32) void wmma_gemm_bf16_tdm(
    const __bf16* __restrict__ Abf, const float* __restrict__ Bt,
    const float* __restrict__ bias, float* __restrict__ C, int N, int K)
{
  __shared__ float tileB[16 * 516];          // 33KB, padded rows
  const int lane = threadIdx.x;
  const int hf   = lane >> 4;
  const int m16  = lane & 15;
  const int n0   = blockIdx.x * 16;
  const int n    = n0 + m16;

  // ---- TDM descriptor: 2D tensor [N rows x K cols] fp32, tile [16 x K] ----
  const unsigned lds_off = (unsigned)(size_t)(&tileB[0]);
  const unsigned long long ga =
      (unsigned long long)(size_t)Bt + (unsigned long long)n0 * (unsigned)K * 4ull;

  v4u g0;
  g0[0] = 1u;                                            // count = 1 valid D#
  g0[1] = lds_off;                                       // lds_addr
  g0[2] = (unsigned)(ga & 0xFFFFFFFFull);                // global_addr[31:0]
  g0[3] = (unsigned)((ga >> 32) & 0x1FFFFFFull)          // global_addr[56:32]
          | (2u << 30);                                  // type = 2 (image)
  v8i g1;
  g1[0] = (2 << 16)            // data_size = 4B
        | (1 << 20)            // pad_enable
        | (7 << 22)            // pad_interval = 256 DWORDs
        | (1 << 25);           // pad_amount   = 2 DWORDs
  g1[1] = (K & 0xFFFF) << 16;                              // tensor_dim0[15:0]
  g1[2] = ((K >> 16) & 0xFFFF) | ((N & 0xFFFF) << 16);     // dim0 hi | tensor_dim1 lo
  g1[3] = ((N >> 16) & 0xFFFF) | ((K & 0xFFFF) << 16);     // dim1 hi | tile_dim0 = K
  g1[4] = 16;                                              // tile_dim1 = 16 rows
  g1[5] = K;                                               // tensor_dim0_stride lo
  g1[6] = 0;
  g1[7] = 0;
  v4i gz = {0, 0, 0, 0};
#if defined(__clang_major__) && (__clang_major__ >= 23)
  v8i gz8 = {0, 0, 0, 0, 0, 0, 0, 0};
  __builtin_amdgcn_tensor_load_to_lds(g0, g1, gz, gz, gz8, 0);
#else
  __builtin_amdgcn_tensor_load_to_lds(g0, g1, gz, gz, 0);
#endif
  __builtin_amdgcn_s_wait_tensorcnt(0);

  v8f acc[8] = {};
  for (int k = 0; k < K; k += 32) {
    // B fragment from LDS (fp32 -> bf16); +2 accounts for the mid-row pad
    const float* brow = &tileB[m16 * 516 + k + ((k >= 256) ? 2 : 0)];
    v16bf bfrag;
#pragma unroll
    for (int j = 0; j < 8; ++j) {
      const int kk = ((j < 4) ? 2 * j : 16 + 2 * (j - 4)) + 8 * hf;
      bfrag[2 * j]     = (__bf16)brow[kk];
      bfrag[2 * j + 1] = (__bf16)brow[kk + 1];
    }
#pragma unroll
    for (int t = 0; t < 8; ++t) {
      const __bf16* arow = Abf + (size_t)(t * 16 + m16) * K + k;
      v8bf lo = *(const v8bf*)(arow + 8 * hf);        // K = k+8hf   .. +7
      v8bf hi = *(const v8bf*)(arow + 16 + 8 * hf);   // K = k+16+8hf.. +7
      v16bf afrag = __builtin_shufflevector(lo, hi, 0, 1, 2, 3, 4, 5, 6, 7,
                                            8, 9, 10, 11, 12, 13, 14, 15);
      acc[t] = __builtin_amdgcn_wmma_f32_16x16x32_bf16(false, afrag, false, bfrag,
                                                       (short)0, acc[t], false, false);
    }
  }
  if (n < N) {
    const float b0 = bias ? bias[n] : 0.0f;
#pragma unroll
    for (int t = 0; t < 8; ++t)
#pragma unroll
      for (int v = 0; v < 8; ++v)
        C[(size_t)(t * 16 + v + 8 * hf) * N + n] = acc[t][v] + b0;
  }
}

// ---------------------------------------------------------------------------
// LSTM pointwise: i,f,g,o -> c1, h1
// ---------------------------------------------------------------------------
__global__ void lstm_kernel(const float* __restrict__ gates,
                            const float* __restrict__ c0,
                            float* __restrict__ h1, float* __restrict__ c1)
{
  const int idx = blockIdx.x * blockDim.x + threadIdx.x;
  if (idx >= B_ * H_) return;
  const int b = idx / H_, h = idx % H_;
  const float* g = gates + (size_t)b * 4 * H_;
  const float gi = 1.0f / (1.0f + expf(-g[h]));
  const float gf = 1.0f / (1.0f + expf(-g[H_ + h]));
  const float gg = tanhf(g[2 * H_ + h]);
  const float go = 1.0f / (1.0f + expf(-g[3 * H_ + h]));
  const float c  = gf * c0[idx] + gi * gg;
  c1[idx] = c;
  h1[idx] = go * tanhf(c);
}

// ---------------------------------------------------------------------------
// Attention (one block of 512 threads = 16 waves per batch row).
// ---------------------------------------------------------------------------
__global__ __launch_bounds__(512) void attn_kernel(
    const float* __restrict__ dec_fea, const float* __restrict__ states_features,
    const float* __restrict__ states, const float* __restrict__ source_mask,
    const float* __restrict__ coverage, const float* __restrict__ v_att,
    const float* __restrict__ w_c, float* __restrict__ attn_out,
    float* __restrict__ cov_out, float* __restrict__ ctx)
{
  __shared__ float e_lds[512];
  __shared__ float red[512];
  __shared__ float df[H_];
  __shared__ float va[H_];
  const int b = blockIdx.x;
  const int tid = threadIdx.x;
  const int wave = tid >> 5;
  const int lane = tid & 31;
  const float wc = w_c[0];

  df[tid] = dec_fea[b * H_ + tid];
  va[tid] = v_att[tid];
  __syncthreads();

  for (int s = wave; s < S_; s += 16) {
    const float cov = coverage[b * S_ + s];
    const float* sf = states_features + ((size_t)b * S_ + s) * H_;
    float p = 0.0f;
    for (int h = lane; h < H_; h += 32)
      p += va[h] * tanhf(sf[h] + df[h] + wc * cov);
#pragma unroll
    for (int off = 16; off > 0; off >>= 1) p += __shfl_xor(p, off, 32);
    if (lane == 0) e_lds[s] = p;
  }
  __syncthreads();

  float em = -3.0e38f, msk = 0.0f;
  if (tid < S_) {
    msk = source_mask[b * S_ + tid];
    em  = (msk > 0.0f) ? e_lds[tid] : -1.0e9f;
  }
  red[tid] = em; __syncthreads();
  for (int off = 256; off > 0; off >>= 1) {
    if (tid < off) red[tid] = fmaxf(red[tid], red[tid + off]);
    __syncthreads();
  }
  const float mx = red[0]; __syncthreads();

  const float p = (tid < S_) ? expf(em - mx) : 0.0f;
  red[tid] = p; __syncthreads();
  for (int off = 256; off > 0; off >>= 1) {
    if (tid < off) red[tid] += red[tid + off];
    __syncthreads();
  }
  const float s1 = red[0]; __syncthreads();

  const float a0 = (p / s1) * msk;
  red[tid] = a0; __syncthreads();
  for (int off = 256; off > 0; off >>= 1) {
    if (tid < off) red[tid] += red[tid + off];
    __syncthreads();
  }
  const float s2 = red[0]; __syncthreads();

  const float a = a0 / (s2 + 1e-12f);
  e_lds[tid] = a;
  if (tid < S_) {
    attn_out[b * S_ + tid] = a;
    cov_out[b * S_ + tid]  = coverage[b * S_ + tid] + a;
  }
  __syncthreads();

  // ctx: 512 threads == H, coalesced reads of states
  {
    const float* st = states + (size_t)b * S_ * H_ + tid;
    float acc = 0.0f;
    for (int s = 0; s < S_; ++s) acc += e_lds[s] * st[(size_t)s * H_];
    ctx[b * H_ + tid] = acc;
  }
}

// ---------------------------------------------------------------------------
__global__ __launch_bounds__(32) void pgen_kernel(
    const float* __restrict__ dec_out, const float* __restrict__ W_pg,
    const float* __restrict__ b_pg, float* __restrict__ p_gen)
{
  const int b = blockIdx.x;
  const int lane = threadIdx.x;
  float acc = 0.0f;
  for (int h = lane; h < H_; h += 32) acc += dec_out[b * H_ + h] * W_pg[h];
#pragma unroll
  for (int off = 16; off > 0; off >>= 1) acc += __shfl_xor(acc, off, 32);
  if (lane == 0) p_gen[b] = 1.0f / (1.0f + expf(-(acc + b_pg[0])));
}

// ---------------------------------------------------------------------------
// Vocab softmax (one block per row; logits stay L2-resident across passes).
// ---------------------------------------------------------------------------
__global__ __launch_bounds__(1024) void vocab_softmax_kernel(
    const float* __restrict__ logits, const float* __restrict__ p_gen,
    float* __restrict__ final_out)
{
  __shared__ float red[1024];
  const int b = blockIdx.x;
  const int tid = threadIdx.x;
  const float* row = logits + (size_t)b * V_;

  float mx = -3.0e38f;
  for (int v = tid; v < V_; v += 1024) mx = fmaxf(mx, row[v]);
  red[tid] = mx; __syncthreads();
  for (int off = 512; off > 0; off >>= 1) {
    if (tid < off) red[tid] = fmaxf(red[tid], red[tid + off]);
    __syncthreads();
  }
  mx = red[0]; __syncthreads();

  float sum = 0.0f;
  for (int v = tid; v < V_; v += 1024) sum += expf(row[v] - mx);
  red[tid] = sum; __syncthreads();
  for (int off = 512; off > 0; off >>= 1) {
    if (tid < off) red[tid] += red[tid + off];
    __syncthreads();
  }
  sum = red[0];

  const float scale = p_gen[b] / sum;
  float* out = final_out + (size_t)b * VEXT_;
  for (int v = tid; v < V_; v += 1024) out[v] = expf(row[v] - mx) * scale;
  for (int v = V_ + tid; v < VEXT_; v += 1024) out[v] = 0.0f;
}

// ---------------------------------------------------------------------------
// Pointer-copy scatter: final[b, src_id[b,s]] += (1-p_gen[b]) * attn[b,s]
// ---------------------------------------------------------------------------
__global__ void scatter_kernel(
    const float* __restrict__ attn, const float* __restrict__ p_gen,
    const int* __restrict__ source_ids, float* __restrict__ final_out)
{
  const int b = blockIdx.x;
  const float w = 1.0f - p_gen[b];
  for (int s = threadIdx.x; s < S_; s += blockDim.x) {
    const int id = source_ids[b * S_ + s];
    atomicAdd(&final_out[(size_t)b * VEXT_ + id], w * attn[b * S_ + s]);
  }
}

// ===========================================================================
extern "C" void kernel_launch(void* const* d_in, const int* in_sizes, int n_in,
                              void* d_out, int out_size, void* d_ws, size_t ws_size,
                              hipStream_t stream)
{
  const float* input_emb       = (const float*)d_in[0];
  const float* input_feed      = (const float*)d_in[1];
  const float* hidden          = (const float*)d_in[2];
  const float* context         = (const float*)d_in[3];
  const float* states          = (const float*)d_in[4];
  const float* states_features = (const float*)d_in[5];
  const float* source_mask     = (const float*)d_in[6];
  const float* coverage        = (const float*)d_in[7];
  const int*   source_ids      = (const int*)d_in[8];
  const float* W_ic  = (const float*)d_in[9];
  const float* b_ic  = (const float*)d_in[10];
  const float* W_ih  = (const float*)d_in[11];
  const float* W_hh  = (const float*)d_in[12];
  const float* b_ih  = (const float*)d_in[13];
  const float* b_hh  = (const float*)d_in[14];
  const float* W_dec = (const float*)d_in[15];
  const float* b_att = (const float*)d_in[16];
  const float* v_att = (const float*)d_in[17];
  const float* w_c   = (const float*)d_in[18];
  const float* W_out = (const float*)d_in[19];
  const float* b_out = (const float*)d_in[20];
  const float* W_pg  = (const float*)d_in[21];
  const float* b_pg  = (const float*)d_in[22];
  const float* W_v   = (const float*)d_in[23];
  const float* b_v   = (const float*)d_in[24];

  // Output layout (reference return order, flat):
  float* out       = (float*)d_out;
  float* out_final = out;                           // [B, VEXT]
  float* out_pgen  = out + (size_t)B_ * VEXT_;      // [B, 1]
  float* out_attn  = out_pgen + B_;                 // [B, S]
  float* out_cov   = out_attn + (size_t)B_ * S_;    // [B, S]
  float* out_h1    = out_cov + (size_t)B_ * S_;     // [1, B, H]
  float* out_c1    = out_h1 + (size_t)B_ * H_;      // [1, B, H]
  float* out_dec   = out_c1 + (size_t)B_ * H_;      // [B, 1, H]

  // Workspace layout (~27.7 MB)
  float*  ws        = (float*)d_ws;
  float*  ws_x      = ws;                             // [B, E]
  float*  ws_gates  = ws_x + (size_t)B_ * E_;         // [B, 4H]
  float*  ws_decfea = ws_gates + (size_t)B_ * 4 * H_; // [B, H]
  float*  ws_ctx    = ws_decfea + (size_t)B_ * H_;    // [B, H]
  float*  ws_logits = ws_ctx + (size_t)B_ * H_;       // [B, V]
  __bf16* ws_abf    = (__bf16*)(ws_logits + (size_t)B_ * V_); // [B, H] bf16

  const dim3 wblk(32);

  // 1) x = [emb|feed] @ W_ic^T + b_ic           (split concat into two K-blocks)
  wmma_gemm_f32<<<dim3(E_ / 16), wblk, 0, stream>>>(
      input_emb, W_ic, H_ + E_, E_,
      input_feed, W_ic + E_, H_ + E_, H_,
      b_ic, nullptr, ws_x, E_);

  // 2) gates = x @ W_ih^T + h0 @ W_hh^T + b_ih + b_hh
  wmma_gemm_f32<<<dim3((4 * H_) / 16), wblk, 0, stream>>>(
      ws_x, W_ih, E_, E_,
      hidden, W_hh, H_, H_,
      b_ih, b_hh, ws_gates, 4 * H_);

  // 3) LSTM pointwise -> h1, c1 (written straight into d_out)
  lstm_kernel<<<dim3((B_ * H_ + 255) / 256), dim3(256), 0, stream>>>(
      ws_gates, context, out_h1, out_c1);

  // 4) dec_fea = h1 @ W_dec^T + b_att
  wmma_gemm_f32<<<dim3(H_ / 16), wblk, 0, stream>>>(
      out_h1, W_dec, H_, H_,
      nullptr, nullptr, 0, 0,
      b_att, nullptr, ws_decfea, H_);

  // 5) attention: e/tanh/mask/softmax/coverage/ctx fused per batch row
  attn_kernel<<<dim3(B_), dim3(512), 0, stream>>>(
      ws_decfea, states_features, states, source_mask, coverage,
      v_att, w_c, out_attn, out_cov, ws_ctx);

  // 6) dec_out = [h1|ctx] @ W_out^T + b_out
  wmma_gemm_f32<<<dim3(H_ / 16), wblk, 0, stream>>>(
      out_h1, W_out, 2 * H_, H_,
      ws_ctx, W_out + H_, 2 * H_, H_,
      b_out, nullptr, out_dec, H_);

  // 7) p_gen = sigmoid(dec_out @ W_pg^T + b_pg)
  pgen_kernel<<<dim3(B_), wblk, 0, stream>>>(out_dec, W_pg, b_pg, out_pgen);

  // 8a) A -> bf16 once (L2-resident for all vocab-GEMM blocks)
  cvt_bf16_kernel<<<dim3((B_ * H_ + 255) / 256), dim3(256), 0, stream>>>(
      out_dec, ws_abf, B_ * H_);

  // 8b) logits = dec_out @ W_v^T + b_v  (bf16 WMMA, TDM-staged B tiles)
  wmma_gemm_bf16_tdm<<<dim3((V_ + 15) / 16), wblk, 0, stream>>>(
      ws_abf, W_v, b_v, ws_logits, V_, H_);

  // 9) final[:, :V] = p_gen * softmax(logits); final[:, V:] = 0
  vocab_softmax_kernel<<<dim3(B_), dim3(1024), 0, stream>>>(
      ws_logits, out_pgen, out_final);

  // 10) final[b, src_ids[b,s]] += (1-p_gen)*attn
  scatter_kernel<<<dim3(B_), dim3(256), 0, stream>>>(
      out_attn, out_pgen, source_ids, out_final);
}
